// DecoderRNN_68805376082025
// MI455X (gfx1250) — compile-verified, matrix operand order checked by reading, not run on hardware
//
#include <hip/hip_runtime.h>
#include <math.h>

// ---------------------------------------------------------------------------
// DecoderRNN for MI455X (gfx1250), persistent single-WGP design.
//
//  - One workgroup, 1024 threads (32 wave32s) runs all 256 decode steps.
//  - All mat-vecs use V_WMMA_F32_16X16X32_F16 in "matvec mode": A holds 16
//    weight rows x 32 K, B broadcasts the activation vector across N.
//  - Recurrent state, gates, softmax, attention scratch live in LDS.
//  - Weights converted once to f16 in d_ws; they stay resident in the 192MB
//    L2 across all 256 steps (~5.4MB total), so HBM is touched once.
//
// d_ws layout (needs ~5.5 MB):
//   [f16]  W_ih(3x1024x256) | W_hh(3x1024x256) | att_W(256x256) |
//          encT(256x1024)   | w1(256x512) | w2(256x256) | w3pad(64x256) |
//          xs(256x256 teacher-forced embedded inputs)
//   [f32]  VOut(256x1024)
// ---------------------------------------------------------------------------

typedef __attribute__((ext_vector_type(16))) _Float16 v16h;
typedef __attribute__((ext_vector_type(8)))  _Float16 v8h;
typedef __attribute__((ext_vector_type(8)))  float    v8f;

constexpr int kH     = 256;
constexpr int k4H    = 1024;
constexpr int kS     = 1024;
constexpr int kV     = 47;
constexpr int kSteps = 256;   // T-1

// ws offsets in _Float16 units
constexpr size_t OFF_WIH  = 0;
constexpr size_t OFF_WHH  = OFF_WIH + (size_t)3 * k4H * kH;
constexpr size_t OFF_ATTW = OFF_WHH + (size_t)3 * k4H * kH;
constexpr size_t OFF_ENCT = OFF_ATTW + (size_t)kH * kH;
constexpr size_t OFF_W1   = OFF_ENCT + (size_t)kH * kS;
constexpr size_t OFF_W2   = OFF_W1 + (size_t)kH * 2 * kH;
constexpr size_t OFF_W3   = OFF_W2 + (size_t)kH * kH;
constexpr size_t OFF_XS   = OFF_W3 + (size_t)64 * kH;
constexpr size_t HALFS_TOTAL = OFF_XS + (size_t)kSteps * kH;

union HU { v16h v; v8h p[2]; };

__device__ __forceinline__ float fast_tanh(float x) {
#if __has_builtin(__builtin_amdgcn_tanhf)
  return __builtin_amdgcn_tanhf(x);      // gfx1250 v_tanh_f32 (TRANS)
#else
  return tanhf(x);
#endif
}
__device__ __forceinline__ float sigmoidf(float x) {
  return 1.0f / (1.0f + __expf(-x));
}

__device__ __forceinline__ v8f wmma16(v16h a, v16h b, v8f c) {
  // (neg_a, A, neg_b, B, c_mod, C, reuse_a, reuse_b)
  return __builtin_amdgcn_wmma_f32_16x16x32_f16(false, a, false, b,
                                                (short)0, c, false, false);
}

// A fragment: 16 rows (row0..row0+15) x 32 K of row-major f16 W (stride ldw).
// 16-bit A layout: lanes 0-15 hold K {K0..K0+7, K0+16..K0+23},
//                  lanes 16-31 hold K {K0+8..K0+15, K0+24..K0+31}.
__device__ __forceinline__ v16h load_a_frag(const _Float16* __restrict__ W,
                                            int ldw, int row0, int K0, int lane) {
  int m  = row0 + (lane & 15);
  int hi = (lane >> 4) & 1;
  const _Float16* r = W + (size_t)m * ldw + K0 + hi * 8;
  HU u;
  u.p[0] = *(const v8h*)(r);
  u.p[1] = *(const v8h*)(r + 16);
  return u.v;
}

// B fragment broadcasting vector x across all 16 N columns (same per-lane K
// packing as A, mirrored for the B 32x16 layout).
__device__ __forceinline__ v16h load_b_bcast(const _Float16* x, int K0, int lane) {
  int hi = (lane >> 4) & 1;
  const _Float16* r = x + K0 + hi * 8;
  HU u;
  u.p[0] = *(const v8h*)(r);
  u.p[1] = *(const v8h*)(r + 16);
  return u.v;
}

// One wave computes 32 output rows [row0, row0+32) of y += W(rows x KLEN) @ x.
template <int KLEN>
__device__ __forceinline__ void wave_matvec32(const _Float16* __restrict__ W,
                                              int ldw, const _Float16* x,
                                              int row0, int lane,
                                              v8f& acc0, v8f& acc1) {
#pragma unroll
  for (int K0 = 0; K0 < KLEN; K0 += 32) {
    v16h b  = load_b_bcast(x, K0, lane);
    v16h a0 = load_a_frag(W, ldw, row0,      K0, lane);
    acc0 = wmma16(a0, b, acc0);
    v16h a1 = load_a_frag(W, ldw, row0 + 16, K0, lane);
    acc1 = wmma16(a1, b, acc1);
  }
}

// D layout (f32 16x16): lanes 0-15 VGPR r -> (M=r, N=lane); lanes 16-31 -> M=r+8.
// All N columns are identical (B broadcast), so lanes 0 and 16 publish results.
__device__ __forceinline__ void store_acc32(float* dst, int row0, int lane,
                                            v8f a0, v8f a1) {
  if (lane == 0) {
#pragma unroll
    for (int r = 0; r < 8; ++r) { dst[row0 + r]      = a0[r]; dst[row0 + 16 + r] = a1[r]; }
  } else if (lane == 16) {
#pragma unroll
    for (int r = 0; r < 8; ++r) { dst[row0 + 8 + r]  = a0[r]; dst[row0 + 24 + r] = a1[r]; }
  }
}

// ------------------------------ setup kernels ------------------------------

__global__ void cvt_f16(_Float16* dst, const float* src, int n) {
  int i = blockIdx.x * 256 + threadIdx.x;
  if (i < n) dst[i] = (_Float16)src[i];
}

__global__ void pad_w3(_Float16* dst, const float* src) {  // 47x256 -> 64x256
  int i = blockIdx.x * 256 + threadIdx.x;
  int r = i >> 8, c = i & 255;
  dst[i] = (r < kV) ? (_Float16)src[r * kH + c] : (_Float16)0.0f;
}

__global__ void build_encT(_Float16* dst, const float* enc) {  // dst[h][s]=enc[s][h]
  int i = blockIdx.x * 256 + threadIdx.x;   // i = h*1024 + s
  int h = i >> 10, s = i & 1023;
  dst[i] = (_Float16)enc[(size_t)s * kH + h];
}

__global__ void build_xs(_Float16* dst, const float* emb, const int* Y) {
  int i = blockIdx.x * 256 + threadIdx.x;   // i = t*256 + k
  int t = i >> 8, k = i & 255;
  dst[i] = (_Float16)emb[(size_t)Y[t] * kH + k];
}

// VOut[i][s] = sum_k att_V[i][k] * encT[k][s]   (coalesced over s)
__global__ void build_vout(float* vout, const float* attV, const _Float16* encT) {
  int i = blockIdx.x >> 2;
  int s = ((blockIdx.x & 3) << 8) + threadIdx.x;
  const float* av = attV + (size_t)i * kH;
  float acc = 0.f;
#pragma unroll 4
  for (int k = 0; k < kH; ++k)
    acc = fmaf(av[k], (float)encT[(size_t)k * kS + s], acc);
  vout[(size_t)i * kS + s] = acc;
}

// ----------------------------- persistent kernel ---------------------------

struct DecArgs {
  const _Float16 *wih, *whh, *attw, *enct, *w1, *w2, *w3, *xs;
  const float *vout;
  const float *h0, *c0;
  const float *bih1, *bhh1, *bih2, *bhh2, *bih3, *bhh3;
  const float *avec, *ab, *b1, *b2, *b3;
  float* out;
};

__global__ __launch_bounds__(1024, 1) void decoder_persistent(DecArgs A) {
  __shared__ __align__(16) _Float16 x16[kH];        // current layer input (f16)
  __shared__ __align__(16) _Float16 hprev16[3][kH]; // recurrent h (f16, prev step)
  __shared__ __align__(16) _Float16 a16[kS];        // softmax alpha (f16)
  __shared__ __align__(16) _Float16 vbuf16[2 * kH]; // [h2 ; context]
  __shared__ __align__(16) _Float16 y116[kH];       // MLP hidden (f16)
  __shared__ float cfb[3][kH];                      // cell state (fp32)
  __shared__ float h2f[kH];                         // h2 (fp32)
  __shared__ float gbuf[k4H];                       // gate / matvec results
  __shared__ float red[kS];                         // softmax reduction
  __shared__ float wsb[kH];                         // att_W@h2 + att_b
  __shared__ float s_avec[kH], s_ab[kH];

  const int tid  = threadIdx.x;
  const int lane = tid & 31;
  const int wave = tid >> 5;

  const float* bih[3] = {A.bih1, A.bih2, A.bih3};
  const float* bhh[3] = {A.bhh1, A.bhh2, A.bhh3};

  if (tid < kH) {
#pragma unroll
    for (int l = 0; l < 3; ++l) {
      float hv = A.h0[l * kH + tid];
      cfb[l][tid]     = A.c0[l * kH + tid];
      hprev16[l][tid] = (_Float16)hv;
    }
    s_avec[tid] = A.avec[tid];
    s_ab[tid]   = A.ab[tid];
  }
  __syncthreads();

  for (int t = 0; t < kSteps; ++t) {
    // ---- teacher-forced input x(t) ----
    if (tid < kH) x16[tid] = A.xs[(size_t)t * kH + tid];
    __syncthreads();

    // ---- 3 LSTM layers ----
    for (int l = 0; l < 3; ++l) {
      const _Float16* Wih = A.wih + (size_t)l * k4H * kH;
      const _Float16* Whh = A.whh + (size_t)l * k4H * kH;
      v8f acc0 = {}, acc1 = {};
      int row0 = wave * 32;                       // 32 waves cover 1024 gate rows
      wave_matvec32<kH>(Wih, kH, x16,        row0, lane, acc0, acc1);
      wave_matvec32<kH>(Whh, kH, hprev16[l], row0, lane, acc0, acc1);
      store_acc32(gbuf, row0, lane, acc0, acc1);
      __syncthreads();
      if (tid < kH) {                              // gate order i,f,g,o
        float gi = gbuf[tid]            + bih[l][tid]            + bhh[l][tid];
        float gf = gbuf[kH + tid]       + bih[l][kH + tid]       + bhh[l][kH + tid];
        float gg = gbuf[2 * kH + tid]   + bih[l][2 * kH + tid]   + bhh[l][2 * kH + tid];
        float go = gbuf[3 * kH + tid]   + bih[l][3 * kH + tid]   + bhh[l][3 * kH + tid];
        float cn = sigmoidf(gf) * cfb[l][tid] + sigmoidf(gi) * fast_tanh(gg);
        float hn = sigmoidf(go) * fast_tanh(cn);
        cfb[l][tid]     = cn;
        hprev16[l][tid] = (_Float16)hn;           // recurrent input for step t+1
        x16[tid]        = (_Float16)hn;           // input to next layer
        if (l == 2) h2f[tid] = hn;
      }
      __syncthreads();
    }

    // ---- attention: WS = att_W @ h2 ----
    {
      v8f acc0 = {}, acc1 = {};
      if (wave < 8) {
        wave_matvec32<kH>(A.attw, kH, x16, wave * 32, lane, acc0, acc1);
        store_acc32(gbuf, wave * 32, lane, acc0, acc1);
      }
      __syncthreads();
      if (tid < kH) wsb[tid] = gbuf[tid] + s_ab[tid];
      __syncthreads();
    }

    // ---- e[s] = sum_j avec[j] * tanh(wsb[j] + VOut[j][s]) ----
    float es = 0.f;
    {
      const float* vc = A.vout + tid;              // column tid, coalesced
#pragma unroll 4
      for (int j = 0; j < kH; ++j)
        es = fmaf(s_avec[j], fast_tanh(wsb[j] + vc[(size_t)j * kS]), es);
    }
    red[tid] = es;
    __syncthreads();
    for (int off = 512; off > 0; off >>= 1) {      // max-reduce
      if (tid < off) red[tid] = fmaxf(red[tid], red[tid + off]);
      __syncthreads();
    }
    float mx = red[0];
    __syncthreads();
    float ex = __expf(es - mx);
    red[tid] = ex;
    __syncthreads();
    for (int off = 512; off > 0; off >>= 1) {      // sum-reduce
      if (tid < off) red[tid] += red[tid + off];
      __syncthreads();
    }
    a16[tid] = (_Float16)(ex / red[0]);
    __syncthreads();

    // ---- context = encT @ alpha  (256 rows, K=1024) ----
    {
      v8f acc0 = {}, acc1 = {};
      if (wave < 8) {
        wave_matvec32<kS>(A.enct, kS, a16, wave * 32, lane, acc0, acc1);
        store_acc32(gbuf, wave * 32, lane, acc0, acc1);
      }
      __syncthreads();
      if (tid < kH) {
        vbuf16[tid]      = (_Float16)h2f[tid];
        vbuf16[kH + tid] = (_Float16)gbuf[tid];
      }
      __syncthreads();
    }

    // ---- MLP1: relu(w1 @ v + b1), 256 rows, K=512 ----
    {
      v8f acc0 = {}, acc1 = {};
      if (wave < 8) {
        wave_matvec32<2 * kH>(A.w1, 2 * kH, vbuf16, wave * 32, lane, acc0, acc1);
        store_acc32(gbuf, wave * 32, lane, acc0, acc1);
      }
      __syncthreads();
      if (tid < kH) y116[tid] = (_Float16)fmaxf(gbuf[tid] + A.b1[tid], 0.f);
      __syncthreads();
    }
    // ---- MLP2: relu(w2 @ y1 + b2), 256 rows, K=256 ----
    {
      v8f acc0 = {}, acc1 = {};
      if (wave < 8) {
        wave_matvec32<kH>(A.w2, kH, y116, wave * 32, lane, acc0, acc1);
        store_acc32(gbuf, wave * 32, lane, acc0, acc1);
      }
      __syncthreads();
      if (tid < kH) y116[tid] = (_Float16)fmaxf(gbuf[tid] + A.b2[tid], 0.f);
      __syncthreads();
    }
    // ---- MLP3: out = w3 @ y2 + b3, 64 padded rows, K=256 ----
    {
      v8f acc0 = {}, acc1 = {};
      if (wave < 2) {
        wave_matvec32<kH>(A.w3, kH, y116, wave * 32, lane, acc0, acc1);
        store_acc32(gbuf, wave * 32, lane, acc0, acc1);
      }
      __syncthreads();
      if (tid < kV) A.out[(size_t)t * kV + tid] = gbuf[tid] + A.b3[tid];
      __syncthreads();
    }
  }
}

// ------------------------------- launcher ----------------------------------

extern "C" void kernel_launch(void* const* d_in, const int* in_sizes, int n_in,
                              void* d_out, int out_size, void* d_ws, size_t ws_size,
                              hipStream_t stream) {
  (void)in_sizes; (void)n_in; (void)out_size; (void)ws_size;  // ws needs ~5.5MB

  _Float16* wsH = (_Float16*)d_ws;
  float* vout = (float*)((char*)d_ws + HALFS_TOTAL * sizeof(_Float16));

  // one-time (but idempotent, replayed per call) weight preparation
  for (int l = 0; l < 3; ++l) {
    cvt_f16<<<1024, 256, 0, stream>>>(wsH + OFF_WIH + (size_t)l * k4H * kH,
                                      (const float*)d_in[5 + 4 * l], k4H * kH);
    cvt_f16<<<1024, 256, 0, stream>>>(wsH + OFF_WHH + (size_t)l * k4H * kH,
                                      (const float*)d_in[6 + 4 * l], k4H * kH);
  }
  cvt_f16<<<256,  256, 0, stream>>>(wsH + OFF_ATTW, (const float*)d_in[18], kH * kH);
  cvt_f16<<<512,  256, 0, stream>>>(wsH + OFF_W1,   (const float*)d_in[21], kH * 2 * kH);
  cvt_f16<<<256,  256, 0, stream>>>(wsH + OFF_W2,   (const float*)d_in[23], kH * kH);
  pad_w3 <<<64,   256, 0, stream>>>(wsH + OFF_W3,   (const float*)d_in[25]);
  build_encT<<<1024, 256, 0, stream>>>(wsH + OFF_ENCT, (const float*)d_in[3]);
  build_xs  <<<256,  256, 0, stream>>>(wsH + OFF_XS, (const float*)d_in[4],
                                       (const int*)d_in[0]);
  build_vout<<<1024, 256, 0, stream>>>(vout, (const float*)d_in[19], wsH + OFF_ENCT);

  DecArgs A;
  A.wih  = wsH + OFF_WIH;  A.whh = wsH + OFF_WHH;  A.attw = wsH + OFF_ATTW;
  A.enct = wsH + OFF_ENCT; A.w1  = wsH + OFF_W1;   A.w2   = wsH + OFF_W2;
  A.w3   = wsH + OFF_W3;   A.xs  = wsH + OFF_XS;   A.vout = vout;
  A.h0   = (const float*)d_in[1];  A.c0   = (const float*)d_in[2];
  A.bih1 = (const float*)d_in[7];  A.bhh1 = (const float*)d_in[8];
  A.bih2 = (const float*)d_in[11]; A.bhh2 = (const float*)d_in[12];
  A.bih3 = (const float*)d_in[15]; A.bhh3 = (const float*)d_in[16];
  A.avec = (const float*)d_in[17]; A.ab   = (const float*)d_in[20];
  A.b1   = (const float*)d_in[22]; A.b2   = (const float*)d_in[24];
  A.b3   = (const float*)d_in[26];
  A.out  = (float*)d_out;

  decoder_persistent<<<1, 1024, 0, stream>>>(A);
}
